// ViT_model_70729521431224
// MI455X (gfx1250) — matrix operand order repaired and checked by reading, not compiled, not run on hardware
//
#include <hip/hip_runtime.h>

#define B_    8
#define N_    1024
#define C_    256
#define H_    8
#define D_    32
#define HID_  512
#define EPS_  1e-5f

typedef __attribute__((ext_vector_type(16))) __bf16 v16bf;
typedef __attribute__((ext_vector_type(8)))  __bf16 v8bf;
typedef __attribute__((ext_vector_type(8)))  float  v8f;

// CDNA5 async global->LDS DMA (ASYNCcnt-tracked), if this toolchain exposes it.
#if __has_builtin(__builtin_amdgcn_global_load_async_to_lds_b128)
#define ATTN_ASYNC 1
// Builtin prototype (from hipcc diagnostic): (v4i AS1*, v4i AS3*, imm, imm)
typedef int b128vec __attribute__((__vector_size__(16)));
#endif

union V16U { v16bf v; v8bf h[2]; };

// B-operand (32x16, K x N): lane = column, K runs [koff,koff+8) and
// [16+koff,16+koff+8), koff = (lane>=16 ? 8 : 0); caller passes pointer offset by koff.
__device__ inline v16bf ldB16(const __bf16* p) {
  V16U u; u.h[0] = *(const v8bf*)p; u.h[1] = *(const v8bf*)(p + 16); return u.v;
}
// A-operand (16x32, M x K): lane = row, K runs [koff,koff+8) and [16+koff,16+koff+8)
__device__ inline v16bf ldA16(const __bf16* row, int koff) {
  V16U u; u.h[0] = *(const v8bf*)(row + koff);
  u.h[1] = *(const v8bf*)(row + 16 + koff); return u.v;
}
__device__ inline v8f wmma_bf16(v16bf a, v16bf b, v8f c) {
  return __builtin_amdgcn_wmma_f32_16x16x32_bf16(false, a, false, b, (short)0, c,
                                                 false, false);
}

// -------------------------------------------------------------------------
// Kernel 1: per-token 3x3 "same" conv on the 16x16 patch -> Q,K (bf16,
// [b,h,n,32] row-major) and V transposed (bf16, [b,feat(256),n]).
// -------------------------------------------------------------------------
__global__ __launch_bounds__(256) void conv_qkv_k(
    const float* __restrict__ x, const float* __restrict__ qw,
    const float* __restrict__ kw, const float* __restrict__ vw,
    __bf16* __restrict__ qo, __bf16* __restrict__ ko, __bf16* __restrict__ vto) {
  int tok = blockIdx.x;           // b*N + n
  int b = tok >> 10, n = tok & 1023;
  int tid = threadIdx.x, i = tid >> 4, j = tid & 15;
  __shared__ float t[16][17];
  __shared__ float wq[9], wk[9], wv[9];
  t[i][j] = x[(size_t)tok * C_ + tid];
  if (tid < 9) { wq[tid] = qw[tid]; wk[tid] = kw[tid]; wv[tid] = vw[tid]; }
  __syncthreads();
  float sq = 0.f, sk = 0.f, sv = 0.f;
#pragma unroll
  for (int di = -1; di <= 1; ++di)
#pragma unroll
    for (int dj = -1; dj <= 1; ++dj) {
      int ii = i + di, jj = j + dj;
      if (ii >= 0 && ii < 16 && jj >= 0 && jj < 16) {
        float v = t[ii][jj]; int wi = (di + 1) * 3 + (dj + 1);
        sq += wq[wi] * v; sk += wk[wi] * v; sv += wv[wi] * v;
      }
    }
  int h = tid >> 5, d = tid & 31;
  size_t qi = (((size_t)(b * H_ + h)) * N_ + n) * D_ + d;
  qo[qi] = (__bf16)sq;
  ko[qi] = (__bf16)sk;
  vto[((size_t)(b * C_ + tid)) * N_ + n] = (__bf16)sv;
}

// -------------------------------------------------------------------------
// Kernel 2: column sums of V (Σ_m v[b,feat,m]) for the BN-bias constant term
// -------------------------------------------------------------------------
__global__ __launch_bounds__(256) void colsum_k(const __bf16* __restrict__ vt,
                                                float* __restrict__ cs) {
  int tid = threadIdx.x, wv = tid >> 5, lane = tid & 31;
  int row = blockIdx.x * 8 + wv;            // 0 .. B*256-1
  const __bf16* p = vt + (size_t)row * N_;
  float s = 0.f;
  for (int m = lane; m < N_; m += 32) s += (float)p[m];
  for (int off = 16; off > 0; off >>= 1) s += __shfl_xor(s, off, 32);
  if (lane == 0) cs[row] = s;
}

// -------------------------------------------------------------------------
// Kernel 3: fp32 -> bf16 weight conversion
// -------------------------------------------------------------------------
__global__ void cvt_bf16_k(const float* __restrict__ s, __bf16* __restrict__ d,
                           int n) {
  int i = blockIdx.x * 256 + threadIdx.x;
  if (i < n) d[i] = (__bf16)s[i];
}

// -------------------------------------------------------------------------
// Kernel 4: flash attention with fused re-attention (head mix) + BN affine.
// Block = 16 query rows, 8 waves (wave = head). Per 32-key chunk:
//   S = q@k^T (2 WMMA), online softmax, P -> LDS relayout, U += P@Vcat (16 WMMA)
// V chunks are staged with GLOBAL_LOAD_ASYNC_TO_LDS (ASYNCcnt) one chunk
// ahead, overlapping the DMA with the next chunk's S/softmax work.
// Epilogue: o[g] = c1[g]*Σ_h w[g,h]*U_h[:, g*32:(g+1)*32]/l + c0[g]*colsum(v_g)
// -------------------------------------------------------------------------
__global__ __launch_bounds__(256) void attn_k(
    const __bf16* __restrict__ q, const __bf16* __restrict__ k,
    const __bf16* __restrict__ vt, const float* __restrict__ cs,
    const float* __restrict__ rw_g, const float* __restrict__ rb,
    const float* __restrict__ bng, const float* __restrict__ bnb,
    __bf16* __restrict__ o) {
  int b = blockIdx.y, n0 = blockIdx.x * 16;
  int tid = threadIdx.x, w = tid >> 5, lane = tid & 31, lcol = lane & 15;
  int hi8 = (lane & 16) ? 8 : 0;
  int koffA = (lane & 16) ? 8 : 0, koffB = (lane & 16) ? 8 : 0;

  __shared__ __bf16 Vsh[256][40];   // V chunk, padded rows (80B, 16B aligned)
  __shared__ __bf16 Psh[8][16][40]; // per-wave P staging for C->A relayout
  __shared__ float  Osh[16][256];   // head-mixed output accumulator

#pragma unroll
  for (int r = 0; r < 16; ++r) Osh[r][tid] = 0.f;

  const __bf16* qrow = q + (((size_t)(b * H_ + w)) * N_ + n0 + lcol) * D_;
  v16bf qa = ldA16(qrow, koffA);

  v8f U[16];
#pragma unroll
  for (int t = 0; t < 16; ++t)
#pragma unroll
    for (int v = 0; v < 8; ++v) U[t][v] = 0.f;
  float mrun[8], lrun[8];
#pragma unroll
  for (int v = 0; v < 8; ++v) { mrun[v] = -1e30f; lrun[v] = 0.f; }
  const float scale = 0.17677669529663687f;   // 32^-0.5

  const __bf16* kbase = k + ((size_t)(b * H_ + w)) * N_ * D_;
  const __bf16* vbase = vt + (size_t)b * C_ * N_;

#ifdef ATTN_ASYNC
  // Per-lane async DMA: thread 'tid' stages its padded 64B V feature row as
  // 4 x b128; the instruction offset applies to both global and LDS address.
  auto issueV = [&](int m0) {
    const __bf16* vp = vbase + (size_t)tid * N_ + m0;
    auto* g = (__attribute__((address_space(1))) b128vec*)vp;
    auto* l = (__attribute__((address_space(3))) b128vec*)&Vsh[tid][0];
    __builtin_amdgcn_global_load_async_to_lds_b128(g, l, 0, 0);
    __builtin_amdgcn_global_load_async_to_lds_b128(g, l, 16, 0);
    __builtin_amdgcn_global_load_async_to_lds_b128(g, l, 32, 0);
    __builtin_amdgcn_global_load_async_to_lds_b128(g, l, 48, 0);
  };
  issueV(0);   // prime the pipeline
#endif

  for (int mc = 0; mc < N_ / 32; ++mc) {
    int m0 = mc * 32;

    const __bf16* krow = kbase + (size_t)(m0 + lcol) * D_ + koffB;
    v16bf kb0 = ldB16(krow);
    v16bf kb1 = ldB16(krow + 16 * D_);
    if (mc + 1 < N_ / 32) __builtin_prefetch(krow + 32 * D_, 0, 1);

    v8f z = {0.f, 0.f, 0.f, 0.f, 0.f, 0.f, 0.f, 0.f};
    v8f s0 = wmma_bf16(qa, kb0, z);
    v8f s1 = wmma_bf16(qa, kb1, z);

    float p0[8], p1[8], scl[8];
#pragma unroll
    for (int v = 0; v < 8; ++v) {
      s0[v] *= scale; s1[v] *= scale;
      float pm = fmaxf(s0[v], s1[v]);             // row reduce over 16 lanes
      pm = fmaxf(pm, __shfl_xor(pm, 1, 32));
      pm = fmaxf(pm, __shfl_xor(pm, 2, 32));
      pm = fmaxf(pm, __shfl_xor(pm, 4, 32));
      pm = fmaxf(pm, __shfl_xor(pm, 8, 32));
      float mnew = fmaxf(mrun[v], pm);
      scl[v] = __expf(mrun[v] - mnew);
      p0[v] = __expf(s0[v] - mnew);
      p1[v] = __expf(s1[v] - mnew);
      float r = p0[v] + p1[v];
      r += __shfl_xor(r, 1, 32); r += __shfl_xor(r, 2, 32);
      r += __shfl_xor(r, 4, 32); r += __shfl_xor(r, 8, 32);
      lrun[v] = lrun[v] * scl[v] + r;
      mrun[v] = mnew;
    }
#pragma unroll
    for (int t = 0; t < 16; ++t)
#pragma unroll
      for (int v = 0; v < 8; ++v) U[t][v] *= scl[v];

    // C-layout (row = v + hi8, col = lcol / lcol+16) -> LDS
#pragma unroll
    for (int v = 0; v < 8; ++v) {
      Psh[w][v + hi8][lcol]      = (__bf16)p0[v];
      Psh[w][v + hi8][lcol + 16] = (__bf16)p1[v];
    }

#ifdef ATTN_ASYNC
#if __has_builtin(__builtin_amdgcn_s_wait_asynccnt)
    __builtin_amdgcn_s_wait_asynccnt(0);
#else
    asm volatile("s_wait_asynccnt 0" ::: "memory");
#endif
#else
    {  // synchronous fallback staging of this chunk
      const __bf16* vp = vbase + (size_t)tid * N_ + m0;
      *(v8bf*)&Vsh[tid][0]  = *(const v8bf*)(vp);
      *(v8bf*)&Vsh[tid][8]  = *(const v8bf*)(vp + 8);
      *(v8bf*)&Vsh[tid][16] = *(const v8bf*)(vp + 16);
      *(v8bf*)&Vsh[tid][24] = *(const v8bf*)(vp + 24);
    }
#endif
    __syncthreads();   // V chunk + all P tiles visible

    v16bf pa = ldA16(&Psh[w][lcol][0], koffA);
#pragma unroll
    for (int t = 0; t < 16; ++t) {
      v16bf vb = ldB16(&Vsh[t * 16 + lcol][koffB]);
      U[t] = wmma_bf16(pa, vb, U[t]);
    }
    __syncthreads();   // everyone done reading Vsh/Psh

#ifdef ATTN_ASYNC
    if (mc + 1 < N_ / 32) issueV(m0 + 32);   // overlap next DMA with softmax
#endif
  }

  float rl[8];
#pragma unroll
  for (int v = 0; v < 8; ++v) rl[v] = 1.f / lrun[v];
  float rwv[8];
#pragma unroll
  for (int g = 0; g < 8; ++g) rwv[g] = rw_g[g * H_ + w];  // w[g, h=this wave]

#pragma unroll
  for (int t = 0; t < 16; ++t) {                 // feature tile t -> head g=t/2
    float wg = rwv[t >> 1];
#pragma unroll
    for (int v = 0; v < 8; ++v)
      atomicAdd(&Osh[v + hi8][t * 16 + lcol], U[t][v] * rl[v] * wg);
  }
  __syncthreads();

  int g = tid >> 5;
  float c1 = bng[g] * rsqrtf(1.f + EPS_);
  float c0 = rb[g] * c1 + bnb[g];
  float cv = cs[b * C_ + tid];
#pragma unroll
  for (int r = 0; r < 16; ++r)
    o[(((size_t)b * N_) + n0 + r) * C_ + tid] =
        (__bf16)(c1 * Osh[r][tid] + c0 * cv);
}

// -------------------------------------------------------------------------
// Kernel 5: WMMA GEMM  out = A[M,K] @ W[Nn,K]^T (+bias, +residual | GELU)
// wave -> 16x64 output tile; block (8 waves) -> 128x64.
// -------------------------------------------------------------------------
template <bool GELU>
__global__ __launch_bounds__(256) void gemm_k(
    const __bf16* __restrict__ A, const __bf16* __restrict__ W,
    const float* __restrict__ bias, const float* __restrict__ res,
    float* __restrict__ outf, __bf16* __restrict__ outb, int M, int Nn, int K) {
  int tid = threadIdx.x, w = tid >> 5, lane = tid & 31, lcol = lane & 15;
  int koffA = (lane & 16) ? 8 : 0, koffB = (lane & 16) ? 8 : 0;
  int row0 = blockIdx.y * 128 + w * 16;
  int col0 = blockIdx.x * 64;

  v8f acc[4];
#pragma unroll
  for (int jt = 0; jt < 4; ++jt)
#pragma unroll
    for (int v = 0; v < 8; ++v) acc[jt][v] = 0.f;

  const __bf16* arow = A + (size_t)(row0 + lcol) * K;
  for (int k0 = 0; k0 < K; k0 += 32) {
    v16bf a = ldA16(arow + k0, koffA);
    if (k0 + 32 < K) __builtin_prefetch(arow + k0 + 32, 0, 1);
#pragma unroll
    for (int jt = 0; jt < 4; ++jt) {
      const __bf16* wrow =
          W + (size_t)(col0 + jt * 16 + lcol) * K + k0 + koffB;
      v16bf bt = ldB16(wrow);
      acc[jt] = wmma_bf16(a, bt, acc[jt]);
    }
  }

  int rbase = row0 + ((lane & 16) ? 8 : 0);
#pragma unroll
  for (int jt = 0; jt < 4; ++jt) {
    int col = col0 + jt * 16 + lcol;
    float bv = bias[col];
#pragma unroll
    for (int v = 0; v < 8; ++v) {
      int r = rbase + v;
      float val = acc[jt][v] + bv;
      if (GELU) {
        float gg = 0.5f * val * (1.f + erff(val * 0.70710678118654752f));
        outb[(size_t)r * Nn + col] = (__bf16)gg;
      } else {
        outf[(size_t)r * Nn + col] = val + res[(size_t)r * Nn + col];
      }
    }
  }
}

// -------------------------------------------------------------------------
// Kernel 6: LayerNorm over (N,C) jointly per batch, [N,C] affine.
// -------------------------------------------------------------------------
template <bool WBF>
__global__ __launch_bounds__(1024) void ln_k(
    const float* __restrict__ in, const float* __restrict__ g,
    const float* __restrict__ bt, float* __restrict__ outf,
    __bf16* __restrict__ outb) {
  int b = blockIdx.x, tid = threadIdx.x;
  const int NC = N_ * C_;
  const float* p = in + (size_t)b * NC;
  float s = 0.f, s2 = 0.f;
  for (int i = tid; i < NC; i += 1024) { float v = p[i]; s += v; s2 += v * v; }
  __shared__ float r1[1024], r2[1024];
  r1[tid] = s; r2[tid] = s2; __syncthreads();
  for (int off = 512; off > 0; off >>= 1) {
    if (tid < off) { r1[tid] += r1[tid + off]; r2[tid] += r2[tid + off]; }
    __syncthreads();
  }
  float mean = r1[0] * (1.f / NC);
  float var = r2[0] * (1.f / NC) - mean * mean;
  float rstd = rsqrtf(var + EPS_);
  for (int i = tid; i < NC; i += 1024) {
    float v = (p[i] - mean) * rstd * g[i] + bt[i];
    outf[(size_t)b * NC + i] = v;
    if (WBF) outb[(size_t)b * NC + i] = (__bf16)v;
  }
}

// -------------------------------------------------------------------------
extern "C" void kernel_launch(void* const* d_in, const int* in_sizes, int n_in,
                              void* d_out, int out_size, void* d_ws,
                              size_t ws_size, hipStream_t stream) {
  const float* x   = (const float*)d_in[0];
  const float* qw  = (const float*)d_in[1];
  const float* kw  = (const float*)d_in[2];
  const float* vw  = (const float*)d_in[3];
  const float* rw  = (const float*)d_in[4];   // [H,H,1,1] -> [g][h]
  const float* rb  = (const float*)d_in[5];
  const float* bng = (const float*)d_in[6];
  const float* bnb = (const float*)d_in[7];
  const float* pw  = (const float*)d_in[8];   // [C,C]
  const float* pb  = (const float*)d_in[9];
  const float* lng = (const float*)d_in[10];  // [N,C]
  const float* lnb = (const float*)d_in[11];
  const float* w1  = (const float*)d_in[12];  // [HID,C]
  const float* b1  = (const float*)d_in[13];
  const float* w2  = (const float*)d_in[14];  // [C,HID]
  const float* b2  = (const float*)d_in[15];
  float* out = (float*)d_out;

  char* ws = (char*)d_ws;
  size_t off = 0;
  auto alloc = [&](size_t bytes) -> char* {
    char* p = ws + off; off += (bytes + 255) & ~(size_t)255; return p;
  };
  __bf16* qbf   = (__bf16*)alloc((size_t)B_ * H_ * N_ * D_ * 2);   // 4 MB
  __bf16* kbf   = (__bf16*)alloc((size_t)B_ * H_ * N_ * D_ * 2);   // 4 MB
  __bf16* vtbf  = (__bf16*)alloc((size_t)B_ * C_ * N_ * 2);        // 4 MB
  float*  csum  = (float*) alloc((size_t)B_ * C_ * 4);             // 8 KB
  __bf16* obf   = (__bf16*)alloc((size_t)B_ * N_ * C_ * 2);        // 4 MB
  float*  proj  = (float*) alloc((size_t)B_ * N_ * C_ * 4);        // 8 MB
  float*  yf    = (float*) alloc((size_t)B_ * N_ * C_ * 4);        // 8 MB
  __bf16* ybf   = (__bf16*)alloc((size_t)B_ * N_ * C_ * 2);        // 4 MB
  __bf16* hbf   = (__bf16*)alloc((size_t)B_ * N_ * HID_ * 2);      // 8 MB
  float*  f2    = (float*) alloc((size_t)B_ * N_ * C_ * 4);        // 8 MB
  __bf16* pwbf  = (__bf16*)alloc((size_t)C_ * C_ * 2);
  __bf16* w1bf  = (__bf16*)alloc((size_t)HID_ * C_ * 2);
  __bf16* w2bf  = (__bf16*)alloc((size_t)C_ * HID_ * 2);

  // 1) QKV 3x3 convs -> bf16 (Q,K row-major, V transposed)
  conv_qkv_k<<<B_ * N_, 256, 0, stream>>>(x, qw, kw, vw, qbf, kbf, vtbf);
  // 2) V column sums
  colsum_k<<<(B_ * C_) / 8, 256, 0, stream>>>(vtbf, csum);
  // 3) weight conversions
  cvt_bf16_k<<<(C_ * C_) / 256, 256, 0, stream>>>(pw, pwbf, C_ * C_);
  cvt_bf16_k<<<(HID_ * C_) / 256, 256, 0, stream>>>(w1, w1bf, HID_ * C_);
  cvt_bf16_k<<<(C_ * HID_) / 256, 256, 0, stream>>>(w2, w2bf, C_ * HID_);
  // 4) flash attention + re-attention + BN
  attn_k<<<dim3(N_ / 16, B_), 256, 0, stream>>>(qbf, kbf, vtbf, csum, rw, rb,
                                                bng, bnb, obf);
  // 5) proj GEMM + bias + residual(x)
  gemm_k<false><<<dim3(C_ / 64, (B_ * N_) / 128), 256, 0, stream>>>(
      obf, pwbf, pb, x, proj, nullptr, B_ * N_, C_, C_);
  // 6) LayerNorm 1 -> y (f32 + bf16)
  ln_k<true><<<B_, 1024, 0, stream>>>(proj, lng, lnb, yf, ybf);
  // 7) FFN1 + GELU -> h (bf16)
  gemm_k<true><<<dim3(HID_ / 64, (B_ * N_) / 128), 256, 0, stream>>>(
      ybf, w1bf, b1, nullptr, nullptr, hbf, B_ * N_, HID_, C_);
  // 8) FFN2 + bias + residual(y)
  gemm_k<false><<<dim3(C_ / 64, (B_ * N_) / 128), 256, 0, stream>>>(
      hbf, w2bf, b2, yf, f2, nullptr, B_ * N_, C_, HID_);
  // 9) LayerNorm 2 -> output
  ln_k<false><<<B_, 1024, 0, stream>>>(f2, lng, lnb, out, nullptr);
  (void)in_sizes; (void)n_in; (void)out_size; (void)ws_size;
}
